// HybridFusionAlignLoss_29549374996777
// MI455X (gfx1250) — matrix operand order repaired and checked by reading, not compiled, not run on hardware
//
#include <hip/hip_runtime.h>
#include <hip/hip_bf16.h>

typedef __attribute__((ext_vector_type(16))) _Float16 v16h;
typedef __attribute__((ext_vector_type(8)))  _Float16 v8h;
typedef __attribute__((ext_vector_type(8)))  float    v8f;

#define Bsz 256
#define Csz 512
#define Nsz 512
#define EPSF 1e-12f
#define PITCH 520   // halves per LDS row: 512 + 8 pad (bank-spread)

__device__ __forceinline__ float warp_sum32(float v) {
#pragma unroll
  for (int o = 16; o > 0; o >>= 1) v += __shfl_xor(v, o, 32);
  return v;
}

__device__ __forceinline__ float dev_scale(const float* ls) {
  return fminf(fmaxf(__expf(ls[0]), 0.01f), 100.0f);
}

// ---------------------------------------------------------------------------
// K1: normalize clip rows (over C); emit transposed f32 (C,B) and f16 (B,C)
// ---------------------------------------------------------------------------
__global__ void k_clip_prep(const float* __restrict__ clip,
                            float* __restrict__ clip_t,
                            _Float16* __restrict__ clip_h) {
  __shared__ float red[256];
  const int k = blockIdx.x, t = threadIdx.x;
  float v0 = clip[k * Csz + t];
  float v1 = clip[k * Csz + t + 256];
  red[t] = v0 * v0 + v1 * v1;
  __syncthreads();
#pragma unroll
  for (int s = 128; s > 0; s >>= 1) {
    if (t < s) red[t] += red[t + s];
    __syncthreads();
  }
  const float inv = 1.0f / fmaxf(sqrtf(red[0]), EPSF);
  const float n0 = v0 * inv, n1 = v1 * inv;
  clip_t[t * Bsz + k] = n0;
  clip_t[(t + 256) * Bsz + k] = n1;
  clip_h[k * Csz + t] = (_Float16)n0;
  clip_h[k * Csz + t + 256] = (_Float16)n1;
}

// ---------------------------------------------------------------------------
// K2: per-(b,n) inverse L2 norm over C of yolo_feats (B,C,N)
// ---------------------------------------------------------------------------
__global__ void k_yolo_norms(const float* __restrict__ yolo,
                             float* __restrict__ invn) {
  const int b = blockIdx.x >> 1;
  const int n = ((blockIdx.x & 1) << 8) + threadIdx.x;
  const float* p = yolo + (size_t)b * Csz * Nsz + n;
  float ss = 0.f;
  for (int c = 0; c < Csz; ++c) {
    float x = p[c * Nsz];
    ss += x * x;
  }
  invn[b * Nsz + n] = 1.0f / fmaxf(sqrtf(ss), EPSF);
}

// ---------------------------------------------------------------------------
// K3: yolo_global[b] = l2norm( mean_n( yolo_norm[b,:,n] ) )
// ---------------------------------------------------------------------------
__global__ void k_yolo_global(const float* __restrict__ yolo,
                              const float* __restrict__ invn,
                              float* __restrict__ yglob) {
  __shared__ float invn_s[Nsz];
  __shared__ float mean_s[Csz];
  __shared__ float red[256];
  const int b = blockIdx.x, t = threadIdx.x;
  invn_s[t] = invn[b * Nsz + t];
  invn_s[t + 256] = invn[b * Nsz + t + 256];
  __syncthreads();
  const int wave = t >> 5, lane = t & 31;
  const float* base = yolo + (size_t)b * Csz * Nsz;
  for (int c = wave; c < Csz; c += 8) {
    const float* row = base + c * Nsz;
    float s = 0.f;
    for (int j = lane; j < Nsz; j += 32) s += row[j] * invn_s[j];
    s = warp_sum32(s);
    if (lane == 0) mean_s[c] = s * (1.0f / Nsz);
  }
  __syncthreads();
  const float a = mean_s[t], bb = mean_s[t + 256];
  red[t] = a * a + bb * bb;
  __syncthreads();
#pragma unroll
  for (int s = 128; s > 0; s >>= 1) {
    if (t < s) red[t] += red[t + s];
    __syncthreads();
  }
  const float inv = 1.0f / fmaxf(sqrtf(red[0]), EPSF);
  yglob[b * Csz + t] = a * inv;
  yglob[b * Csz + t + 256] = bb * inv;
}

// ---------------------------------------------------------------------------
// K4: sim_global = (yolo_global @ clip_norm^T) * scale   (256x256, tiny)
// ---------------------------------------------------------------------------
__global__ void k_sim_global(const float* __restrict__ yglob,
                             const float* __restrict__ clip_t,
                             const float* __restrict__ ls,
                             float* __restrict__ simg) {
  const int i = blockIdx.x, j = threadIdx.x;
  const float scale = dev_scale(ls);
  const float* yg = yglob + i * Csz;
  float acc = 0.f;
  for (int c = 0; c < Csz; ++c) acc += yg[c] * clip_t[c * Bsz + j];
  simg[i * Bsz + j] = acc * scale;
}

// ---------------------------------------------------------------------------
// K5: big branch. Per block b: flash-style logsumexp/max over n of
//     scale * (yolo_norm[b].T (NxC) @ clip_norm.T (CxK)), via f16 WMMA.
//     8 waves; wave w owns k-tiles 2w, 2w+1. Clip (256KB f16) + A-tile in LDS.
// ---------------------------------------------------------------------------
__device__ __forceinline__ void online_update(v8f acc, float scale,
                                              float& m, float& s) {
  float v[8];
  float tmax = -1e30f;
#pragma unroll
  for (int r = 0; r < 8; ++r) {
    v[r] = acc[r] * scale;
    tmax = fmaxf(tmax, v[r]);
  }
  const float mn = fmaxf(m, tmax);
  float sum = s * __expf(m - mn);
#pragma unroll
  for (int r = 0; r < 8; ++r) sum += __expf(v[r] - mn);
  m = mn;
  s = sum;
}

__global__ void k_spatial(const float* __restrict__ yolo,
                          const float* __restrict__ invn,
                          const _Float16* __restrict__ clip_h,
                          const float* __restrict__ ls,
                          float* __restrict__ sim_soft,
                          float* __restrict__ sim_hard) {
  extern __shared__ char smem[];
  _Float16* clip_lds = (_Float16*)smem;              // 256 * PITCH halves
  _Float16* a_lds    = clip_lds + Bsz * PITCH;       // 16 * PITCH halves
  float*    invn_lds = (float*)(a_lds + 16 * PITCH); // 16 floats

  const int b = blockIdx.x, tid = threadIdx.x;
  const int wave = tid >> 5, lane = tid & 31;
  const float scale = dev_scale(ls);

  // Stage the whole normalized clip matrix (k,c) as f16 into LDS, 16B/thread/op
  for (int idx = tid; idx < (Bsz * Csz) / 8; idx += 256) {
    const int k = idx >> 6;
    const int c8 = (idx & 63) << 3;
    *(v8h*)(clip_lds + k * PITCH + c8) = *(const v8h*)(clip_h + k * Csz + c8);
  }

  const int M   = lane & 15;               // A row / B column within tile
  const int kbA = (lane < 16) ? 0 : 16;    // A-frag K half base
  const int cbB = (lane < 16) ? 0 : 16;    // B-frag K (=c) half base
  const int k0a = wave * 32, k0b = wave * 32 + 16;

  float m0 = -1e30f, s0 = 0.f, m1 = -1e30f, s1 = 0.f;
  const float* ybase = yolo + (size_t)b * Csz * Nsz;

  for (int nt = 0; nt < Nsz / 16; ++nt) {
    const int n0 = nt * 16;
    __syncthreads();                                   // a_lds / clip staging safe
    if (tid < 16) invn_lds[tid] = invn[b * Nsz + n0 + tid];
    __syncthreads();
    // Stage A tile: 16 (n) x 512 (c), normalized, f32 -> f16 LDS rows by n
#pragma unroll 4
    for (int it = 0; it < 32; ++it) {
      const int lin = it * 256 + tid;
      const int nn = lin & 15, c = lin >> 4;
      const float v = ybase[c * Nsz + n0 + nn] * invn_lds[nn];
      a_lds[nn * PITCH + c] = (_Float16)v;
    }
    __syncthreads();

    v8f acc0 = {};
    v8f acc1 = {};
    for (int c0 = 0; c0 < Csz; c0 += 32) {
      // A fragment (16x32 f16): contiguous 16-half K-run per lane
      const _Float16* ap = a_lds + M * PITCH + c0 + kbA;
      const v8h alo = *(const v8h*)ap;
      const v8h ahi = *(const v8h*)(ap + 8);
      const v16h af = __builtin_shufflevector(alo, ahi, 0, 1, 2, 3, 4, 5, 6, 7,
                                              8, 9, 10, 11, 12, 13, 14, 15);
      // B fragment k-tile 0 (32x16 f16): contiguous 16-half K-run per lane
      const _Float16* bp0 = clip_lds + (k0a + M) * PITCH + c0 + cbB;
      const v8h b0l = *(const v8h*)bp0;
      const v8h b0h = *(const v8h*)(bp0 + 8);
      const v16h bf0 = __builtin_shufflevector(b0l, b0h, 0, 1, 2, 3, 4, 5, 6, 7,
                                               8, 9, 10, 11, 12, 13, 14, 15);
      acc0 = __builtin_amdgcn_wmma_f32_16x16x32_f16(false, af, false, bf0,
                                                    (short)0, acc0, false, false);
      // B fragment k-tile 1
      const _Float16* bp1 = clip_lds + (k0b + M) * PITCH + c0 + cbB;
      const v8h b1l = *(const v8h*)bp1;
      const v8h b1h = *(const v8h*)(bp1 + 8);
      const v16h bf1 = __builtin_shufflevector(b1l, b1h, 0, 1, 2, 3, 4, 5, 6, 7,
                                               8, 9, 10, 11, 12, 13, 14, 15);
      acc1 = __builtin_amdgcn_wmma_f32_16x16x32_f16(false, af, false, bf1,
                                                    (short)0, acc1, false, false);
    }
    // lane<16 holds rows 0..7 of the tile, lane+16 holds rows 8..15
    online_update(acc0, scale, m0, s0);
    online_update(acc1, scale, m1, s1);
  }

  // combine the two lane-halves of each column (rows 0..7 with rows 8..15)
  const float om0 = __shfl_xor(m0, 16, 32), os0 = __shfl_xor(s0, 16, 32);
  const float Mm0 = fmaxf(m0, om0);
  const float Ss0 = s0 * __expf(m0 - Mm0) + os0 * __expf(om0 - Mm0);
  const float om1 = __shfl_xor(m1, 16, 32), os1 = __shfl_xor(s1, 16, 32);
  const float Mm1 = fmaxf(m1, om1);
  const float Ss1 = s1 * __expf(m1 - Mm1) + os1 * __expf(om1 - Mm1);
  if (lane < 16) {
    sim_soft[b * Bsz + k0a + lane] = Mm0 + __logf(Ss0);
    sim_hard[b * Bsz + k0a + lane] = Mm0;
    sim_soft[b * Bsz + k0b + lane] = Mm1 + __logf(Ss1);
    sim_hard[b * Bsz + k0b + lane] = Mm1;
  }
}

// ---------------------------------------------------------------------------
// K6: focal cross-entropy per row (6 passes: {global,soft,hard} x {S, S^T})
// ---------------------------------------------------------------------------
__global__ void k_focal(const float* __restrict__ simg,
                        const float* __restrict__ sims,
                        const float* __restrict__ simh,
                        float* __restrict__ focal_out) {
  __shared__ float red[256];
  __shared__ float diag_lp;
  const int row = blockIdx.x;
  const int pass = blockIdx.y;       // 0..5
  const int mat = pass >> 1, tr = pass & 1;
  const float* S = (mat == 0) ? simg : (mat == 1) ? sims : simh;
  const int t = threadIdx.x;
  const float val = tr ? S[t * Bsz + row] : S[row * Bsz + t];

  red[t] = val;
  __syncthreads();
#pragma unroll
  for (int s = 128; s > 0; s >>= 1) {
    if (t < s) red[t] = fmaxf(red[t], red[t + s]);
    __syncthreads();
  }
  const float mx = red[0];
  __syncthreads();

  red[t] = __expf(val - mx);
  __syncthreads();
#pragma unroll
  for (int s = 128; s > 0; s >>= 1) {
    if (t < s) red[t] += red[t + s];
    __syncthreads();
  }
  const float lse = mx + __logf(red[0]);
  __syncthreads();

  const float logp = val - lse;
  if (t == row) diag_lp = logp;
  red[t] = logp;
  __syncthreads();
#pragma unroll
  for (int s = 128; s > 0; s >>= 1) {
    if (t < s) red[t] += red[t + s];
    __syncthreads();
  }
  if (t == 0) {
    const float nll = -diag_lp;
    const float smooth = -red[0] * (1.0f / Bsz);
    const float ce = 0.75f * nll + 0.25f * smooth;   // label smoothing 0.25
    const float pt = __expf(-ce);
    const float om = 1.0f - pt;                      // gamma = 2
    const float W[3] = {0.2f, 0.3f, 0.5f};
    focal_out[pass * Bsz + row] = W[mat] * 0.5f * (1.0f / Bsz) * om * om * ce;
  }
}

// K7: deterministic final reduce of 1536 row contributions -> scalar loss
__global__ void k_final(const float* __restrict__ focal_out,
                        float* __restrict__ out) {
  __shared__ float red[256];
  const int t = threadIdx.x;
  float s = 0.f;
  for (int i = t; i < 6 * Bsz; i += 256) s += focal_out[i];
  red[t] = s;
  __syncthreads();
#pragma unroll
  for (int st = 128; st > 0; st >>= 1) {
    if (t < st) red[t] += red[t + st];
    __syncthreads();
  }
  if (t == 0) out[0] = red[0];
}

extern "C" void kernel_launch(void* const* d_in, const int* in_sizes, int n_in,
                              void* d_out, int out_size, void* d_ws,
                              size_t ws_size, hipStream_t stream) {
  (void)in_sizes; (void)n_in; (void)out_size; (void)ws_size;
  const float* yolo = (const float*)d_in[0];  // (256,512,512)
  const float* clip = (const float*)d_in[1];  // (256,512)
  const float* ls   = (const float*)d_in[2];  // (1,)
  float* out = (float*)d_out;

  char* ws = (char*)d_ws;
  float*    invn   = (float*)(ws + 0);          // 256*512*4 = 524288
  float*    clipt  = (float*)(ws + 524288);     // 512*256*4 = 524288
  _Float16* cliph  = (_Float16*)(ws + 1048576); // 256*512*2 = 262144
  float*    yglob  = (float*)(ws + 1310720);    // 256*512*4 = 524288
  float*    simg   = (float*)(ws + 1835008);    // 256*256*4 = 262144
  float*    sims   = (float*)(ws + 2097152);    // 262144
  float*    simh   = (float*)(ws + 2359296);    // 262144
  float*    foc    = (float*)(ws + 2621440);    // 6*256*4 = 6144
  // total workspace: ~2.5 MB

  k_clip_prep<<<Bsz, 256, 0, stream>>>(clip, clipt, cliph);
  k_yolo_norms<<<(Bsz * Nsz) / 256, 256, 0, stream>>>(yolo, invn);
  k_yolo_global<<<Bsz, 256, 0, stream>>>(yolo, invn, yglob);
  k_sim_global<<<Bsz, 256, 0, stream>>>(yglob, clipt, ls, simg);

  const size_t smem = (size_t)(Bsz * PITCH + 16 * PITCH) * sizeof(_Float16) +
                      16 * sizeof(float);  // 282,944 B < 320 KB/WGP
  k_spatial<<<Bsz, 256, smem, stream>>>(yolo, invn, cliph, ls, sims, simh);

  k_focal<<<dim3(Bsz, 6), 256, 0, stream>>>(simg, sims, simh, foc);
  k_final<<<1, 256, 0, stream>>>(foc, out);
}
